// MambaLMHeadModelwithPosids_24017457119639
// MI455X (gfx1250) — compile-verified, matrix operand order checked by reading, not run on hardware
//
#include <hip/hip_runtime.h>
#include <hip/hip_bf16.h>
#include <math.h>

typedef __bf16 v16bf __attribute__((ext_vector_type(16)));
typedef float  v8f   __attribute__((ext_vector_type(8)));

#define B_    2
#define L_    1024
#define MROWS (B_*L_)          /* 2048 */
#define DM    768
#define DI    1536
#define DS_   16
#define DR    48
#define DRP   64               /* dt rank padded to K%64==0 */
#define DXP   80               /* dt_rank + 2*d_state */
#define XDLD  128              /* x_dbl padded row stride / padded N */
#define DE    384
#define VOCAB 32000
#define NL    6

#define MT 128
#define NT 128
#define KS 64                  /* staged K depth (2 wmma K-steps) */
#define STAGE ((MT + NT) * KS) /* bf16 elements per double-buffer stage */

__device__ __forceinline__ unsigned short f2bf(float f) {
  unsigned int u = __float_as_uint(f);
  u += 0x7FFFu + ((u >> 16) & 1u);           // round-to-nearest-even
  return (unsigned short)(u >> 16);
}
__device__ __forceinline__ float silu_f(float x) {
  return x * (1.0f / (1.0f + __expf(-x)));
}
// low 32 bits of a generic LDS pointer == LDS byte offset (ISA 10.2 aperture);
// ptrtoint also makes the shared object escape so the async stores are honored.
__device__ __forceinline__ unsigned lds_addr(const void* p) {
  return (unsigned)(unsigned long long)p;
}

// ---------------------------------------------------------------------------
// Branch-free bf16 WMMA GEMM with async global->LDS double-buffered pipeline.
//   C[m,n] = sum_k A[m,k] * W[n,k]  (+ epilogue)
// Requirements (guaranteed by caller via padding):
//   gridDim.y*MT == M, gridDim.x*NT == N, K % 64 == 0, W row stride == K.
// mode 0: none.  mode 1: softplus(v + bias[n]).
// ---------------------------------------------------------------------------
__global__ __launch_bounds__(256)
void gemm_wmma_bf16(const unsigned short* __restrict__ A, int lda,
                    const unsigned short* __restrict__ W,
                    const float* __restrict__ bias,
                    float* __restrict__ C, int ldc,
                    int K, int mode)
{
  __shared__ alignas(128) unsigned short smem[2 * STAGE];   // 64 KB

  const int tid   = threadIdx.x;
  const int lane  = tid & 31;
  const int wave  = tid >> 5;            // 0..7
  const int wm    = (wave & 3) * 32;     // wave M offset (32 rows)
  const int wn    = (wave >> 2) * 64;    // wave N offset (64 cols)
  const int m0    = blockIdx.y * MT;
  const int n0    = blockIdx.x * NT;
  const int l16   = lane & 15;
  const int khalf = (lane >> 4) * 16;    // K sub-slot per lane half

  v8f acc[2][4];
#pragma unroll
  for (int s = 0; s < 2; ++s)
#pragma unroll
    for (int t = 0; t < 4; ++t)
#pragma unroll
      for (int r = 0; r < 8; ++r) acc[s][t][r] = 0.0f;

  // issue one 128x64 A tile + 128x64 B tile into stage buffer p (8 async
  // B128 loads per wave, ASYNCcnt-tracked, LDS written directly)
  auto issue_stage = [&](int p, int kk) {
    unsigned short* as_base = &smem[p * STAGE];
    unsigned short* bs_base = as_base + MT * KS;
#pragma unroll
    for (int i = 0; i < 4; ++i) {
      int e  = tid + i * 256;
      int r  = e >> 3;                  // row 0..127
      int c8 = (e & 7) * 8;             // col 0,8,...,56
      const unsigned short* ga = A + (size_t)(m0 + r) * lda + kk + c8;
      const unsigned short* gb = W + (size_t)(n0 + r) * K + kk + c8;
      unsigned la = lds_addr(as_base + r * KS + c8);
      unsigned lb = lds_addr(bs_base + r * KS + c8);
      asm volatile("global_load_async_to_lds_b128 %0, %1, off"
                   :: "v"(la), "v"(ga) : "memory");
      asm volatile("global_load_async_to_lds_b128 %0, %1, off"
                   :: "v"(lb), "v"(gb) : "memory");
    }
  };

  issue_stage(0, 0);

  for (int kk = 0; kk < K; kk += KS) {
    const int p = (kk / KS) & 1;
    if (kk + KS < K) {                   // uniform scalar branch
      issue_stage(p ^ 1, kk + KS);       // overlap fill of next stage
      asm volatile("s_wait_asynccnt 0x8" ::: "memory");  // stage p landed
    } else {
      asm volatile("s_wait_asynccnt 0x0" ::: "memory");
    }
    __syncthreads();                     // stage p visible to all waves

    const unsigned short* As = &smem[p * STAGE];
    const unsigned short* Bs = &smem[p * STAGE + MT * KS];
#pragma unroll
    for (int kh = 0; kh < 2; ++kh) {
      const int kc = kh * 32 + khalf;
      const v16bf a0 = *(const v16bf*)(&As[(wm + l16) * KS + kc]);
      const v16bf a1 = *(const v16bf*)(&As[(wm + 16 + l16) * KS + kc]);
#pragma unroll
      for (int t = 0; t < 4; ++t) {
        const v16bf b = *(const v16bf*)(&Bs[(wn + t * 16 + l16) * KS + kc]);
        acc[0][t] = __builtin_amdgcn_wmma_f32_16x16x32_bf16(
            false, a0, false, b, (short)0, acc[0][t], false, false);
        acc[1][t] = __builtin_amdgcn_wmma_f32_16x16x32_bf16(
            false, a1, false, b, (short)0, acc[1][t], false, false);
      }
    }
    __syncthreads();                     // stage p free for refill at kk+2
  }

  // ---- epilogue + store (C layout: vgpr r -> m = r + 8*(lane/16)) ----
#pragma unroll
  for (int s = 0; s < 2; ++s) {
#pragma unroll
    for (int t = 0; t < 4; ++t) {
      int n = n0 + wn + t * 16 + l16;
      float bv = (mode == 1) ? bias[n] : 0.0f;
#pragma unroll
      for (int r = 0; r < 8; ++r) {
        int m = m0 + wm + s * 16 + r + (lane >> 4) * 8;
        float v = acc[s][t][r];
        if (mode == 1) {
          v += bv;
          v = (v > 20.0f) ? v : log1pf(__expf(v));   // softplus
        }
        C[(size_t)m * ldc + n] = v;
      }
    }
  }
}

// ---------------------------------------------------------------------------
// Embedding concat
// ---------------------------------------------------------------------------
__global__ void embed_kernel(const int* __restrict__ ids, const int* __restrict__ pids,
                             const float* __restrict__ emb, const float* __restrict__ pemb,
                             float* __restrict__ hidden)
{
  int idx = blockIdx.x * blockDim.x + threadIdx.x;
  if (idx >= MROWS * DM) return;
  int row = idx / DM, d = idx % DM;
  float v = (d < DE) ? emb[(size_t)ids[row] * DE + d]
                     : pemb[(size_t)pids[row] * DE + (d - DE)];
  hidden[idx] = v;
}

// ---------------------------------------------------------------------------
// residual += hidden; out_bf16 = rmsnorm(residual) * w   (block per row)
// ---------------------------------------------------------------------------
__global__ __launch_bounds__(256)
void add_rmsnorm_kernel(const float* __restrict__ hidden, float* __restrict__ residual,
                        const float* __restrict__ w, unsigned short* __restrict__ out_bf)
{
  __shared__ float red[256];
  int row = blockIdx.x;
  int tid = threadIdx.x;
  float v[3];
  float ss = 0.0f;
#pragma unroll
  for (int i = 0; i < 3; ++i) {
    int d = tid + i * 256;
    float x = residual[(size_t)row * DM + d] + hidden[(size_t)row * DM + d];
    v[i] = x;
    ss += x * x;
  }
  red[tid] = ss;
  __syncthreads();
  for (int s = 128; s > 0; s >>= 1) {
    if (tid < s) red[tid] += red[tid + s];
    __syncthreads();
  }
  float inv = rsqrtf(red[0] * (1.0f / DM) + 1e-5f);
#pragma unroll
  for (int i = 0; i < 3; ++i) {
    int d = tid + i * 256;
    residual[(size_t)row * DM + d] = v[i];
    out_bf[(size_t)row * DM + d]   = f2bf(v[i] * inv * w[d]);
  }
}

// ---------------------------------------------------------------------------
// Depthwise causal conv1d (k=4) + SiLU over x half of xz; writes f32 + bf16
// ---------------------------------------------------------------------------
__global__ void conv_silu_kernel(const float* __restrict__ xz,
                                 const float* __restrict__ cw, const float* __restrict__ cb,
                                 float* __restrict__ xf, unsigned short* __restrict__ xbf)
{
  int idx = blockIdx.x * blockDim.x + threadIdx.x;
  if (idx >= MROWS * DI) return;
  int d   = idx % DI;
  int row = idx / DI;
  int l = row % L_, b = row / L_;
  float acc = cb[d];
#pragma unroll
  for (int j = 0; j < 4; ++j) {
    int ls = l - 3 + j;
    if (ls >= 0) acc += cw[d * 4 + j] * xz[((size_t)(b * L_ + ls)) * (2 * DI) + d];
  }
  float s = silu_f(acc);
  xf[idx]  = s;
  xbf[idx] = f2bf(s);
}

// dt rank slice of x_dbl -> bf16, K padded 48 -> 64 with zeros
__global__ void dtslice_kernel(const float* __restrict__ xdbl, unsigned short* __restrict__ dtin)
{
  int idx = blockIdx.x * blockDim.x + threadIdx.x;
  if (idx >= MROWS * DRP) return;
  int row = idx / DRP, r = idx % DRP;
  dtin[idx] = (r < DR) ? f2bf(xdbl[(size_t)row * XDLD + r]) : (unsigned short)0;
}

// ---------------------------------------------------------------------------
// Selective scan: one thread per (b, channel); 16 states in registers.
// ---------------------------------------------------------------------------
__global__ __launch_bounds__(256)
void scan_kernel(const float* __restrict__ xconv, const float* __restrict__ dtb,
                 const float* __restrict__ xdbl, const float* __restrict__ A_log,
                 const float* __restrict__ Dvec, float* __restrict__ y)
{
  int c = blockIdx.x * blockDim.x + threadIdx.x;
  if (c >= B_ * DI) return;
  int b = c / DI, d = c % DI;
  float a[DS_], h[DS_];
#pragma unroll
  for (int n = 0; n < DS_; ++n) {
    a[n] = -__expf(A_log[(size_t)d * DS_ + n]);
    h[n] = 0.0f;
  }
  float Dv = Dvec[d];
  for (int l = 0; l < L_; ++l) {
    size_t row = (size_t)(b * L_ + l);
    float x  = xconv[row * DI + d];
    float dt = dtb[row * DI + d];
    const float* xd = xdbl + row * XDLD;
    float yv = 0.0f;
#pragma unroll
    for (int n = 0; n < DS_; ++n) {
      float dA = __expf(dt * a[n]);
      h[n] = dA * h[n] + (dt * xd[DR + n]) * x;
      yv += h[n] * xd[DR + DS_ + n];
    }
    y[row * DI + d] = yv + x * Dv;
  }
}

// y_bf16 = bf16( y * silu(z) )
__global__ void ymul_kernel(const float* __restrict__ y, const float* __restrict__ xz,
                            unsigned short* __restrict__ ybf)
{
  int idx = blockIdx.x * blockDim.x + threadIdx.x;
  if (idx >= MROWS * DI) return;
  int d = idx % DI, row = idx / DI;
  float z = xz[(size_t)row * (2 * DI) + DI + d];
  ybf[idx] = f2bf(y[idx] * silu_f(z));
}

// fp32 -> bf16 bulk convert (dense)
__global__ void cvt_bf16_kernel(const float* __restrict__ src, unsigned short* __restrict__ dst, int n)
{
  int idx = blockIdx.x * blockDim.x + threadIdx.x;
  if (idx < n) dst[idx] = f2bf(src[idx]);
}

// fp32 (R x K) -> bf16 (Rp x Kp), zero padded rows/cols
__global__ void cvt_pad_kernel(const float* __restrict__ src, unsigned short* __restrict__ dst,
                               int R, int K, int Rp, int Kp)
{
  int idx = blockIdx.x * blockDim.x + threadIdx.x;
  if (idx >= Rp * Kp) return;
  int r = idx / Kp, k = idx % Kp;
  dst[idx] = (r < R && k < K) ? f2bf(src[(size_t)r * K + k]) : (unsigned short)0;
}

// ---------------------------------------------------------------------------
extern "C" void kernel_launch(void* const* d_in, const int* in_sizes, int n_in,
                              void* d_out, int out_size, void* d_ws, size_t ws_size,
                              hipStream_t stream)
{
  (void)in_sizes; (void)n_in; (void)out_size; (void)ws_size;
  const int*   ids   = (const int*)d_in[0];
  const int*   pids  = (const int*)d_in[1];
  const float* emb_w = (const float*)d_in[2];
  const float* pemb  = (const float*)d_in[3];
  const float* normw = (const float*)d_in[4];
  const float* inw   = (const float*)d_in[5];
  const float* convw = (const float*)d_in[6];
  const float* convb = (const float*)d_in[7];
  const float* xpw   = (const float*)d_in[8];
  const float* dtw   = (const float*)d_in[9];
  const float* dtbv  = (const float*)d_in[10];
  const float* Alog  = (const float*)d_in[11];
  const float* Dvec  = (const float*)d_in[12];
  const float* outw  = (const float*)d_in[13];
  const float* normf = (const float*)d_in[14];
  float* logits = (float*)d_out;

  // ---- workspace carve-up ----
  char* ws = (char*)d_ws;
  size_t off = 0;
  auto carve = [&](size_t bytes) -> char* {
    char* p = ws + off;
    off += (bytes + 255) & ~(size_t)255;
    return p;
  };
  float* residual = (float*)carve((size_t)MROWS * DM * 4);
  float* hidden   = (float*)carve((size_t)MROWS * DM * 4);
  unsigned short* hs_bf = (unsigned short*)carve((size_t)MROWS * DM * 2);
  float* xz       = (float*)carve((size_t)MROWS * 2 * DI * 4);
  float* xconv_f  = (float*)carve((size_t)MROWS * DI * 4);
  unsigned short* xconv_bf = (unsigned short*)carve((size_t)MROWS * DI * 2);
  float* xdbl     = (float*)carve((size_t)MROWS * XDLD * 4);
  unsigned short* dtin_bf = (unsigned short*)carve((size_t)MROWS * DRP * 2);
  float* dtbuf    = (float*)carve((size_t)MROWS * DI * 4);
  float* ybuf     = (float*)carve((size_t)MROWS * DI * 4);
  unsigned short* ybf = (unsigned short*)carve((size_t)MROWS * DI * 2);
  // weight region: per-layer bf16 weights unioned with emb bf16 (used after layers)
  char* wreg = carve((size_t)VOCAB * DE * 2);    // 24.58 MB, largest user
  unsigned short* w_in_bf  = (unsigned short*)wreg;
  unsigned short* w_xp_bf  = (unsigned short*)(wreg + (size_t)2 * DI * DM * 2);
  unsigned short* w_dt_bf  = (unsigned short*)(wreg + (size_t)2 * DI * DM * 2 + (size_t)XDLD * DI * 2);
  unsigned short* w_out_bf = (unsigned short*)(wreg + (size_t)2 * DI * DM * 2 + (size_t)XDLD * DI * 2 + (size_t)DI * DRP * 2);
  unsigned short* w_emb_bf = (unsigned short*)wreg;

  auto cvt = [&](const float* s, unsigned short* d, int n) {
    cvt_bf16_kernel<<<(n + 255) / 256, 256, 0, stream>>>(s, d, n);
  };
  const dim3 blk(256);

  // ---- init ----
  hipMemsetAsync(residual, 0, (size_t)MROWS * DM * 4, stream);
  embed_kernel<<<(MROWS * DM + 255) / 256, blk, 0, stream>>>(ids, pids, emb_w, pemb, hidden);

  // ---- layers ----
  for (int layer = 0; layer < NL; ++layer) {
    cvt(inw  + (size_t)layer * 2 * DI * DM, w_in_bf,  2 * DI * DM);
    cvt_pad_kernel<<<(XDLD * DI + 255) / 256, blk, 0, stream>>>(
        xpw + (size_t)layer * DXP * DI, w_xp_bf, DXP, DI, XDLD, DI);    // 80x1536 -> 128x1536
    cvt_pad_kernel<<<(DI * DRP + 255) / 256, blk, 0, stream>>>(
        dtw + (size_t)layer * DI * DR, w_dt_bf, DI, DR, DI, DRP);       // 1536x48 -> 1536x64
    cvt(outw + (size_t)layer * DM * DI, w_out_bf, DM * DI);

    add_rmsnorm_kernel<<<MROWS, blk, 0, stream>>>(hidden, residual,
                                                  normw + (size_t)layer * DM, hs_bf);

    // xz = hs @ in_w^T : M=2048, N=3072, K=768
    gemm_wmma_bf16<<<dim3(2 * DI / NT, MROWS / MT), blk, 0, stream>>>(
        hs_bf, DM, w_in_bf, nullptr, xz, 2 * DI, DM, 0);

    conv_silu_kernel<<<(MROWS * DI + 255) / 256, blk, 0, stream>>>(
        xz, convw + (size_t)layer * DI * 4, convb + (size_t)layer * DI, xconv_f, xconv_bf);

    // x_dbl = x @ xp_w^T : N=128 (padded from 80), K=1536
    gemm_wmma_bf16<<<dim3(XDLD / NT, MROWS / MT), blk, 0, stream>>>(
        xconv_bf, DI, w_xp_bf, nullptr, xdbl, XDLD, DI, 0);

    dtslice_kernel<<<(MROWS * DRP + 255) / 256, blk, 0, stream>>>(xdbl, dtin_bf);

    // dt = softplus(dt_in @ dt_w^T + dt_b) : N=1536, K=64 (padded from 48)
    gemm_wmma_bf16<<<dim3(DI / NT, MROWS / MT), blk, 0, stream>>>(
        dtin_bf, DRP, w_dt_bf, dtbv + (size_t)layer * DI, dtbuf, DI, DRP, 1);

    scan_kernel<<<(B_ * DI + 255) / 256, blk, 0, stream>>>(
        xconv_f, dtbuf, xdbl, Alog + (size_t)layer * DI * DS_, Dvec + (size_t)layer * DI, ybuf);

    ymul_kernel<<<(MROWS * DI + 255) / 256, blk, 0, stream>>>(ybuf, xz, ybf);

    // hidden = y @ out_w^T : N=768, K=1536
    gemm_wmma_bf16<<<dim3(DM / NT, MROWS / MT), blk, 0, stream>>>(
        ybf, DI, w_out_bf, nullptr, hidden, DM, DI, 0);
  }

  // ---- final norm + LM head ----
  add_rmsnorm_kernel<<<MROWS, blk, 0, stream>>>(hidden, residual, normf, hs_bf);
  cvt(emb_w, w_emb_bf, VOCAB * DE);
  // logits = hs[:, :384] @ emb^T : M=2048, N=32000, K=384 (lda=768 slices first half)
  gemm_wmma_bf16<<<dim3(VOCAB / NT, MROWS / MT), blk, 0, stream>>>(
      hs_bf, DM, w_emb_bf, nullptr, logits, VOCAB, DE, 0);
}
